// Decoder_27367531610365
// MI455X (gfx1250) — matrix operand order repaired and checked by reading, not compiled, not run on hardware
//
#include <hip/hip_runtime.h>
#include <math.h>

typedef __attribute__((ext_vector_type(16))) __bf16 v16bf;
typedef __attribute__((ext_vector_type(8)))  __bf16 v8bf;
typedef __attribute__((ext_vector_type(8)))  float  v8f;

#define DIMC   1024
#define SEQN   2048
#define NB     2
#define NHEAD  16
#define DHEAD  64
#define NKC    32
#define NR     2
#define NRC    128
#define NCHUNK (NB * NKC)      /* 64 chunks */
#define KVPAD  272             /* 1 null + 256 keys padded to mult of 16 */
#define FFD    4096
#define NROWS  (NB * SEQN)     /* 4096 token rows */
#define NEGF   (-1e30f)

// ---------------------------------------------------------------- helpers
static __device__ __forceinline__ v8f vzero8() {
  v8f r;
#pragma unroll
  for (int i = 0; i < 8; ++i) r[i] = 0.0f;
  return r;
}
// A-fragment 16-bit layout: halves 0..7 = K(base..+7), halves 8..15 = K(base+16..+23)
static __device__ __forceinline__ v16bf ld_frag_gap(const __bf16* p) {
  v8bf lo = *(const v8bf*)(p);
  v8bf hi = *(const v8bf*)(p + 16);
  return __builtin_shufflevector(lo, hi, 0,1,2,3,4,5,6,7,8,9,10,11,12,13,14,15);
}
// B-fragment 16-bit layout: halves 0..7 = K(base..+7), halves 8..15 = K(base+16..+23)
static __device__ __forceinline__ v16bf ld_frag_cont(const __bf16* p) {
  v8bf lo = *(const v8bf*)(p);
  v8bf hi = *(const v8bf*)(p + 16);
  return __builtin_shufflevector(lo, hi, 0,1,2,3,4,5,6,7,8,9,10,11,12,13,14,15);
}
static __device__ __forceinline__ v8f wmma_bf16(v16bf a, v16bf b, v8f c) {
  return __builtin_amdgcn_wmma_f32_16x16x32_bf16(false, a, false, b, (short)0, c, false, false);
}
// async 16B global -> LDS copy (per-lane addresses), tracked by ASYNCcnt
static __device__ __forceinline__ void async_b128(unsigned lds_off, const void* gsrc) {
  asm volatile("global_load_async_to_lds_b128 %0, %1, off"
               :: "v"(lds_off), "v"(gsrc) : "memory");
}
static __device__ __forceinline__ void wait_async0() {
  asm volatile("s_wait_asynccnt 0" ::: "memory");
}
// pack 8+8 contiguous fp32 into a v16bf A-fragment register
static __device__ __forceinline__ v16bf pack16(const float* p0, const float* p1) {
  float4 a = *(const float4*)(p0);     float4 b = *(const float4*)(p0 + 4);
  float4 c = *(const float4*)(p1);     float4 d = *(const float4*)(p1 + 4);
  v16bf r;
  r[0]=(__bf16)a.x;  r[1]=(__bf16)a.y;  r[2]=(__bf16)a.z;  r[3]=(__bf16)a.w;
  r[4]=(__bf16)b.x;  r[5]=(__bf16)b.y;  r[6]=(__bf16)b.z;  r[7]=(__bf16)b.w;
  r[8]=(__bf16)c.x;  r[9]=(__bf16)c.y;  r[10]=(__bf16)c.z; r[11]=(__bf16)c.w;
  r[12]=(__bf16)d.x; r[13]=(__bf16)d.y; r[14]=(__bf16)d.z; r[15]=(__bf16)d.w;
  return r;
}

// ---------------------------------------------------------------- rmsnorm (f32 in; f32 OR bf16 out)
__global__ __launch_bounds__(256) void k_rmsnorm(const float* __restrict__ x,
                                                 const float* __restrict__ g,
                                                 float* __restrict__ outf,
                                                 __bf16* __restrict__ outb) {
  __shared__ float red[256];
  int row = blockIdx.x, tid = threadIdx.x;
  const float* p = x + (size_t)row * DIMC;
  float ss = 0.f;
  for (int e = tid; e < DIMC; e += 256) { float v = p[e]; ss += v * v; }
  red[tid] = ss; __syncthreads();
  for (int s = 128; s > 0; s >>= 1) { if (tid < s) red[tid] += red[tid + s]; __syncthreads(); }
  float n = sqrtf(red[0] * (1.0f / (float)DIMC));
  float inv = 1.0f / fmaxf(n, 1e-8f);
  for (int e = tid; e < DIMC; e += 256) {
    float v = p[e] * inv * g[e];
    if (outb) outb[(size_t)row * DIMC + e] = (__bf16)v;
    else      outf[(size_t)row * DIMC + e] = v;
  }
}

// ---------------------------------------------------------------- f32 -> bf16 bulk convert
__global__ __launch_bounds__(256) void k_cvt(const float* __restrict__ s,
                                             __bf16* __restrict__ d, int n) {
  int i = blockIdx.x * 256 + threadIdx.x;
  if (i < n) d[i] = (__bf16)s[i];
}

// ---------------------------------------------------------------- weight transpose+convert: BT[n][k] = bf16(B[k][n])
__global__ __launch_bounds__(256) void k_transp(const float* __restrict__ B, int K, int N,
                                                __bf16* __restrict__ BT) {
  __shared__ float t[32][33];
  int kb = blockIdx.x * 32, nb = blockIdx.y * 32;
  int tx = threadIdx.x & 31, ty = threadIdx.x >> 5;   // 32 x 8
#pragma unroll
  for (int r = 0; r < 32; r += 8)
    t[ty + r][tx] = B[(size_t)(kb + ty + r) * N + nb + tx];
  __syncthreads();
#pragma unroll
  for (int r = 0; r < 32; r += 8)
    BT[(size_t)(nb + ty + r) * K + kb + tx] = (__bf16)t[tx][ty + r];
}

// ---------------------------------------------------------------- RoPE (in place, first ROT=32 dims of each head)
__global__ __launch_bounds__(256) void k_rope(float* __restrict__ buf, int rowstride,
                                              int seqlen, float scale) {
  int row = blockIdx.x;
  int pos = row % seqlen;
  float* p = buf + (size_t)row * rowstride;
  int tid = threadIdx.x;
  int head = tid >> 4, i = tid & 15;
  float inv = __powf(10000.0f, -(float)(2 * i) * (1.0f / 32.0f));
  float ang = (float)pos * inv;
  float cs = __cosf(ang), sn = __sinf(ang);
  float x1 = p[head * 64 + i]      * scale;
  float x2 = p[head * 64 + i + 16] * scale;
  p[head * 64 + i]      = x1 * cs - x2 * sn;
  p[head * 64 + i + 16] = x2 * cs + x1 * sn;
  if (scale != 1.0f) {
    for (int e = tid; e < NHEAD * 32; e += 256) {
      int h = e >> 5, d = 32 + (e & 31);
      p[h * 64 + d] *= scale;
    }
  }
}

// ---------------------------------------------------------------- cross-attn query gather (+scale, +RoPE row 0 @pos126)
__global__ __launch_bounds__(256) void k_cross_q(const float* __restrict__ qfull,
                                                 float* __restrict__ qc) {
  int row = blockIdx.x;                 // (b*32 + c)*64 + i
  int i = row & 63;
  int c = (row >> 6) & 31;
  int b = row >> 11;
  int t = c * 64 + i + 63;
  float* dst = qc + (size_t)row * DIMC;
  int tid = threadIdx.x;
  if (t < SEQN) {
    const float* src = qfull + ((size_t)b * SEQN + t) * DIMC;
    for (int e = tid; e < DIMC; e += 256) dst[e] = src[e] * 0.125f;
  } else {
    for (int e = tid; e < DIMC; e += 256) dst[e] = 0.f;
  }
  __syncthreads();
  if (i == 0 && t < SEQN) {
    int head = tid >> 4, ii = tid & 15;
    float inv = __powf(10000.0f, -(float)(2 * ii) * (1.0f / 32.0f));
    float ang = 126.0f * inv;
    float cs = __cosf(ang), sn = __sinf(ang);
    float x1 = dst[head * 64 + ii], x2 = dst[head * 64 + ii + 16];
    dst[head * 64 + ii]      = x1 * cs - x2 * sn;
    dst[head * 64 + ii + 16] = x2 * cs + x1 * sn;
  }
}

// ---------------------------------------------------------------- cross-attn KV assembly: null + roped K + pad
__global__ __launch_bounds__(256) void k_cross_kv(const float* __restrict__ ctxkv,
                                                  const float* __restrict__ canull,
                                                  float* __restrict__ kck,
                                                  float* __restrict__ kcv) {
  int row = blockIdx.x;                 // chunk*KVPAD + j
  int chunk = row / KVPAD, j = row % KVPAD;
  float* dk = kck + (size_t)row * DIMC;
  float* dv = kcv + (size_t)row * DIMC;
  int tid = threadIdx.x;
  if (j == 0) {
    for (int e = tid; e < DIMC; e += 256) { dk[e] = canull[e]; dv[e] = canull[DIMC + e]; }
  } else if (j <= NR * NRC) {
    const float* s = ctxkv + ((size_t)chunk * (NR * NRC) + (j - 1)) * (2 * DIMC);
    for (int e = tid; e < DIMC; e += 256) { dk[e] = s[e]; dv[e] = s[DIMC + e]; }
    __syncthreads();
    int pos = (j - 1) & (NRC - 1);
    int head = tid >> 4, i = tid & 15;
    float inv = __powf(10000.0f, -(float)(2 * i) * (1.0f / 32.0f));
    float ang = (float)pos * inv;
    float cs = __cosf(ang), sn = __sinf(ang);
    float x1 = dk[head * 64 + i], x2 = dk[head * 64 + i + 16];
    dk[head * 64 + i]      = x1 * cs - x2 * sn;
    dk[head * 64 + i + 16] = x2 * cs + x1 * sn;
  } else {
    for (int e = tid; e < DIMC; e += 256) { dk[e] = 0.f; dv[e] = 0.f; }
  }
}

// ---------------------------------------------------------------- shifted residual scatter-add of cross projection
__global__ __launch_bounds__(256) void k_scatter(float* __restrict__ x,
                                                 const float* __restrict__ pj) {
  int row = blockIdx.x;                 // b*SEQN + t
  int t = row & (SEQN - 1);
  int b = row >> 11;
  if (t < 63) return;
  const float* s = pj + ((size_t)b * SEQN + (t - 63)) * DIMC;
  float* d = x + (size_t)row * DIMC;
  for (int e = threadIdx.x; e < DIMC; e += 256) d[e] += s[e];
}

// ---------------------------------------------------------------- WMMA bf16 GEMM with async-LDS tile staging
//   A : bf16 [M][K] row-major        BT : bf16 [N][K] row-major (pre-transposed weights)
//   C = A@B (+bias)(+gelu)(+res); flags: 1 bias, 2 gelu, 4 residual(f32), 8 store bf16
#define GLDS 40   /* padded LDS row stride in halves: 80B rows, conflict-free, 16B aligned */
__global__ __launch_bounds__(128) void k_gemm(const __bf16* __restrict__ A, int lda,
                                              const __bf16* __restrict__ BT, int ldbt,
                                              void* __restrict__ Cout, int ldc,
                                              int M, int N, int K,
                                              const float* __restrict__ bias,
                                              const float* __restrict__ res,
                                              int flags) {
  __shared__ __bf16 sA[64 * GLDS];
  __shared__ __bf16 sB[64 * GLDS];
  int tid = threadIdx.x;
  int lane = tid & 31, wave = tid >> 5;
  int half = lane >> 4, l15 = lane & 15;
  int m0 = blockIdx.y * 64, n0 = blockIdx.x * 64;
  int wm = (wave >> 1) * 32, wn = (wave & 1) * 32;
  v8f acc[2][2];
#pragma unroll
  for (int a = 0; a < 2; ++a)
#pragma unroll
    for (int b = 0; b < 2; ++b) acc[a][b] = vzero8();

  // thread -> two 16B chunks per operand tile: chunk c -> row c>>2, 8-half group c&3
  int ca0 = tid, ca1 = tid + 128;
  unsigned dA0 = (unsigned)(size_t)(void*)&sA[(ca0 >> 2) * GLDS + (ca0 & 3) * 8];
  unsigned dA1 = (unsigned)(size_t)(void*)&sA[(ca1 >> 2) * GLDS + (ca1 & 3) * 8];
  unsigned dB0 = (unsigned)(size_t)(void*)&sB[(ca0 >> 2) * GLDS + (ca0 & 3) * 8];
  unsigned dB1 = (unsigned)(size_t)(void*)&sB[(ca1 >> 2) * GLDS + (ca1 & 3) * 8];
  const __bf16* gA0 = A  + (size_t)(m0 + (ca0 >> 2)) * lda  + (ca0 & 3) * 8;
  const __bf16* gA1 = A  + (size_t)(m0 + (ca1 >> 2)) * lda  + (ca1 & 3) * 8;
  const __bf16* gB0 = BT + (size_t)(n0 + (ca0 >> 2)) * ldbt + (ca0 & 3) * 8;
  const __bf16* gB1 = BT + (size_t)(n0 + (ca1 >> 2)) * ldbt + (ca1 & 3) * 8;

  for (int k0 = 0; k0 < K; k0 += 32) {
    __syncthreads();                       // prior tile fully consumed
    async_b128(dA0, gA0 + k0);
    async_b128(dA1, gA1 + k0);
    async_b128(dB0, gB0 + k0);
    async_b128(dB1, gB1 + k0);
    if (k0 + 32 < K) {                     // prefetch next k-tile into L2/WGP$
      __builtin_prefetch(gA0 + k0 + 32, 0, 0);
      __builtin_prefetch(gB0 + k0 + 32, 0, 0);
    }
    wait_async0();
    __syncthreads();                       // all waves' copies visible

    v16bf af[2], bf[2];
#pragma unroll
    for (int mt = 0; mt < 2; ++mt)
      af[mt] = ld_frag_gap(&sA[(wm + mt * 16 + l15) * GLDS + half * 8]);
#pragma unroll
    for (int nt = 0; nt < 2; ++nt)
      bf[nt] = ld_frag_cont(&sB[(wn + nt * 16 + l15) * GLDS + half * 8]);
#pragma unroll
    for (int mt = 0; mt < 2; ++mt)
#pragma unroll
      for (int nt = 0; nt < 2; ++nt)
        acc[mt][nt] = wmma_bf16(af[mt], bf[nt], acc[mt][nt]);
  }
#pragma unroll
  for (int mt = 0; mt < 2; ++mt)
#pragma unroll
    for (int nt = 0; nt < 2; ++nt) {
      int row = m0 + wm + mt * 16 + half * 8;
      int col = n0 + wn + nt * 16 + l15;
#pragma unroll
      for (int r = 0; r < 8; ++r) {
        float v = acc[mt][nt][r];
        if (flags & 1) v += bias[col];
        if (flags & 2) v = 0.5f * v * (1.0f + erff(v * 0.70710678118f));  // exact gelu
        size_t idx = (size_t)(row + r) * ldc + col;
        if (flags & 4) v += res[idx];
        if (flags & 8) ((__bf16*)Cout)[idx] = (__bf16)v;
        else           ((float*)Cout)[idx]  = v;
      }
    }
}

// ---------------------------------------------------------------- WMMA flash attention (one wave / 16-query tile)
__global__ __launch_bounds__(32) void k_flash(const float* __restrict__ Q, int qstride,
                                              const float* __restrict__ Kb,
                                              const float* __restrict__ Vb, int kvstride,
                                              __bf16* __restrict__ O, int ostride,
                                              int q_per_seq, int kv_len, int kv_rows,
                                              int causal,
                                              const unsigned char* __restrict__ mask,
                                              int mask_per_seq) {
  __shared__ __bf16 lK[32 * 72];   // [key][dim]
  __shared__ __bf16 lV[64 * 40];   // [dim][key] (V transposed)
  __shared__ __bf16 lP[16 * 40];   // [qrow][key]
  int lane = threadIdx.x;
  int half = lane >> 4, l15 = lane & 15;
  int q0 = blockIdx.x * 16;
  int head = blockIdx.y;
  int seq = blockIdx.z;

  const float* qbase = Q  + ((size_t)seq * q_per_seq) * qstride + head * DHEAD;
  const float* kbase = Kb + ((size_t)seq * kv_rows) * kvstride + head * DHEAD;
  const float* vbase = Vb + ((size_t)seq * kv_rows) * kvstride + head * DHEAD;

  const float* qr = qbase + (size_t)(q0 + l15) * qstride + half * 8;
  v16bf qa[2];
  qa[0] = pack16(qr,      qr + 16);
  qa[1] = pack16(qr + 32, qr + 48);

  v8f o[4];
  float m_run[8], l_run[8];
#pragma unroll
  for (int t = 0; t < 4; ++t) o[t] = vzero8();
#pragma unroll
  for (int r = 0; r < 8; ++r) { m_run[r] = NEGF; l_run[r] = 0.f; }

  int jmax = causal ? ((q0 + 16) < kv_len ? (q0 + 16) : kv_len) : kv_len;
  for (int j0 = 0; j0 < jmax; j0 += 32) {
    int kr = j0 + lane;
    int krc = kr < kv_rows ? kr : 0;
    const float* krow = kbase + (size_t)krc * kvstride;
    const float* vrow = vbase + (size_t)krc * kvstride;
#pragma unroll
    for (int d0 = 0; d0 < DHEAD; d0 += 4) {
      float4 k4 = *(const float4*)(krow + d0);
      __bf16* dk = &lK[lane * 72 + d0];
      dk[0]=(__bf16)k4.x; dk[1]=(__bf16)k4.y; dk[2]=(__bf16)k4.z; dk[3]=(__bf16)k4.w;
      float4 v4 = *(const float4*)(vrow + d0);
      lV[(d0 + 0) * 40 + lane] = (__bf16)v4.x;
      lV[(d0 + 1) * 40 + lane] = (__bf16)v4.y;
      lV[(d0 + 2) * 40 + lane] = (__bf16)v4.z;
      lV[(d0 + 3) * 40 + lane] = (__bf16)v4.w;
    }
    __syncthreads();

    v8f s[2];
#pragma unroll
    for (int n = 0; n < 2; ++n) {
      v8f a = vzero8();
#pragma unroll
      for (int ks = 0; ks < 2; ++ks) {
        v16bf bf = ld_frag_cont(&lK[(n * 16 + l15) * 72 + ks * 32 + half * 8]);
        a = wmma_bf16(qa[ks], bf, a);
      }
      s[n] = a;
    }

    float p[2][8];
    float mx[8];
#pragma unroll
    for (int r = 0; r < 8; ++r) mx[r] = NEGF;
#pragma unroll
    for (int n = 0; n < 2; ++n) {
      int jp = j0 + n * 16 + l15;
      bool okc = jp < kv_len;
      if (!causal && mask && okc && jp > 0)
        okc = mask[(size_t)seq * mask_per_seq + (jp - 1)] != 0;
#pragma unroll
      for (int r = 0; r < 8; ++r) {
        int qp = q0 + half * 8 + r;
        bool ok = okc && (!causal || jp <= qp);
        float v = ok ? s[n][r] : NEGF;
        p[n][r] = v;
        mx[r] = fmaxf(mx[r], v);
      }
    }
#pragma unroll
    for (int r = 0; r < 8; ++r) {
      float m = mx[r];
      m = fmaxf(m, __shfl_xor(m, 1));
      m = fmaxf(m, __shfl_xor(m, 2));
      m = fmaxf(m, __shfl_xor(m, 4));
      m = fmaxf(m, __shfl_xor(m, 8));
      float newm  = fmaxf(m_run[r], m);
      float alpha = __expf(m_run[r] - newm);
      float p0 = __expf(p[0][r] - newm);
      float p1 = __expf(p[1][r] - newm);
      p[0][r] = p0; p[1][r] = p1;
      float rs = p0 + p1;
      rs += __shfl_xor(rs, 1);
      rs += __shfl_xor(rs, 2);
      rs += __shfl_xor(rs, 4);
      rs += __shfl_xor(rs, 8);
      l_run[r] = l_run[r] * alpha + rs;
      m_run[r] = newm;
#pragma unroll
      for (int t = 0; t < 4; ++t) o[t][r] *= alpha;
    }
#pragma unroll
    for (int n = 0; n < 2; ++n)
#pragma unroll
      for (int r = 0; r < 8; ++r)
        lP[(half * 8 + r) * 40 + n * 16 + l15] = (__bf16)p[n][r];
    __syncthreads();
    v16bf pa = ld_frag_gap(&lP[l15 * 40 + half * 8]);
#pragma unroll
    for (int t = 0; t < 4; ++t) {
      v16bf bv = ld_frag_cont(&lV[(t * 16 + l15) * 40 + half * 8]);
      o[t] = wmma_bf16(pa, bv, o[t]);
    }
    __syncthreads();
  }

#pragma unroll
  for (int t = 0; t < 4; ++t)
#pragma unroll
    for (int r = 0; r < 8; ++r) {
      int row = q0 + half * 8 + r;
      float den = l_run[r];
      float val = o[t][r] / (den > 0.f ? den : 1.f);
      O[((size_t)seq * q_per_seq + row) * ostride + head * DHEAD + t * 16 + l15] = (__bf16)val;
    }
}

// ---------------------------------------------------------------- host orchestration
extern "C" void kernel_launch(void* const* d_in, const int* in_sizes, int n_in,
                              void* d_out, int out_size, void* d_ws, size_t ws_size,
                              hipStream_t stream) {
  (void)in_sizes; (void)n_in; (void)out_size; (void)ws_size;
  const float* x_in   = (const float*)d_in[0];
  const float* retr   = (const float*)d_in[1];
  const unsigned char* cmask = (const unsigned char*)d_in[2];
  const float* sa_g   = (const float*)d_in[3];
  const float* sa_wq  = (const float*)d_in[4];
  const float* sa_wkv = (const float*)d_in[5];
  const float* sa_wo  = (const float*)d_in[6];
  const float* sa_bo  = (const float*)d_in[7];
  const float* ca_g   = (const float*)d_in[8];
  const float* ca_wq  = (const float*)d_in[9];
  const float* ca_wkv = (const float*)d_in[10];
  const float* ca_wo  = (const float*)d_in[11];
  const float* ca_bo  = (const float*)d_in[12];
  const float* ca_nul = (const float*)d_in[13];
  const float* ff_g   = (const float*)d_in[14];
  const float* ff_w1  = (const float*)d_in[15];
  const float* ff_b1  = (const float*)d_in[16];
  const float* ff_w2  = (const float*)d_in[17];
  const float* ff_b2  = (const float*)d_in[18];
  const float* out_g  = (const float*)d_in[19];

  char* wsp = (char*)d_ws;
  auto alloc = [&](size_t bytes) -> void* {
    void* p = (void*)wsp; wsp += (bytes + 255) & ~(size_t)255; return p;
  };
  const size_t CTX = (size_t)NCHUNK * NR * NRC;       // 16384 ctx rows
  float*  xb   = (float*) alloc(sizeof(float)  * NROWS * DIMC);
  float*  qb   = (float*) alloc(sizeof(float)  * NROWS * DIMC);
  float*  kvb  = (float*) alloc(sizeof(float)  * NROWS * 2 * DIMC);
  float*  qc   = (float*) alloc(sizeof(float)  * NROWS * DIMC);
  float*  ckv  = (float*) alloc(sizeof(float)  * CTX * 2 * DIMC);
  float*  kck  = (float*) alloc(sizeof(float)  * NCHUNK * KVPAD * DIMC);
  float*  kcv  = (float*) alloc(sizeof(float)  * NCHUNK * KVPAD * DIMC);
  float*  pj   = (float*) alloc(sizeof(float)  * NROWS * DIMC);
  __bf16* hbf  = (__bf16*)alloc(sizeof(__bf16) * NROWS * DIMC);
  __bf16* abf  = (__bf16*)alloc(sizeof(__bf16) * NROWS * DIMC);
  __bf16* ffbf = (__bf16*)alloc(sizeof(__bf16) * NROWS * FFD);
  __bf16* retbf= (__bf16*)alloc(sizeof(__bf16) * CTX * DIMC);
  __bf16* wtb  = (__bf16*)alloc(sizeof(__bf16) * (size_t)FFD * DIMC);  // max weight

  hipMemcpyAsync(xb, x_in, sizeof(float) * (size_t)NROWS * DIMC,
                 hipMemcpyDeviceToDevice, stream);
  { int n = (int)(CTX * DIMC);
    k_cvt<<<(n + 255) / 256, 256, 0, stream>>>(retr, retbf, n); }

  for (int l = 0; l < 2; ++l) {
    const float* wq  = sa_wq  + (size_t)l * DIMC * DIMC;
    const float* wkv = sa_wkv + (size_t)l * DIMC * 2 * DIMC;
    const float* wo  = sa_wo  + (size_t)l * DIMC * DIMC;
    const float* bo  = sa_bo  + (size_t)l * DIMC;
    const float* cwq  = ca_wq  + (size_t)l * DIMC * DIMC;
    const float* cwkv = ca_wkv + (size_t)l * DIMC * 2 * DIMC;
    const float* cwo  = ca_wo  + (size_t)l * DIMC * DIMC;
    const float* cbo  = ca_bo  + (size_t)l * DIMC;
    const float* cnul = ca_nul + (size_t)l * 2 * DIMC;
    const float* w1 = ff_w1 + (size_t)l * DIMC * FFD;
    const float* b1 = ff_b1 + (size_t)l * FFD;
    const float* w2 = ff_w2 + (size_t)l * FFD * DIMC;
    const float* b2 = ff_b2 + (size_t)l * DIMC;

    // ---- self attention ----
    k_rmsnorm<<<NROWS, 256, 0, stream>>>(xb, sa_g + (size_t)l * DIMC, nullptr, hbf);
    k_transp<<<dim3(DIMC/32, DIMC/32), 256, 0, stream>>>(wq, DIMC, DIMC, wtb);
    k_gemm<<<dim3(DIMC/64, NROWS/64), 128, 0, stream>>>(hbf, DIMC, wtb, DIMC, qb, DIMC,
        NROWS, DIMC, DIMC, nullptr, nullptr, 0);
    k_transp<<<dim3(DIMC/32, 2*DIMC/32), 256, 0, stream>>>(wkv, DIMC, 2*DIMC, wtb);
    k_gemm<<<dim3(2*DIMC/64, NROWS/64), 128, 0, stream>>>(hbf, DIMC, wtb, DIMC, kvb, 2*DIMC,
        NROWS, 2*DIMC, DIMC, nullptr, nullptr, 0);
    k_rope<<<NROWS, 256, 0, stream>>>(qb, DIMC, SEQN, 0.125f);
    k_rope<<<NROWS, 256, 0, stream>>>(kvb, 2*DIMC, SEQN, 1.0f);
    k_flash<<<dim3(SEQN/16, NHEAD, NB), 32, 0, stream>>>(qb, DIMC, kvb, kvb + DIMC, 2*DIMC,
        abf, DIMC, SEQN, SEQN, SEQN, 1, nullptr, 0);
    k_transp<<<dim3(DIMC/32, DIMC/32), 256, 0, stream>>>(wo, DIMC, DIMC, wtb);
    k_gemm<<<dim3(DIMC/64, NROWS/64), 128, 0, stream>>>(abf, DIMC, wtb, DIMC, xb, DIMC,
        NROWS, DIMC, DIMC, bo, xb, 1 | 4);

    // ---- chunked cross attention ----
    k_rmsnorm<<<NROWS, 256, 0, stream>>>(xb, ca_g + (size_t)l * DIMC, nullptr, hbf);
    k_transp<<<dim3(DIMC/32, DIMC/32), 256, 0, stream>>>(cwq, DIMC, DIMC, wtb);
    k_gemm<<<dim3(DIMC/64, NROWS/64), 128, 0, stream>>>(hbf, DIMC, wtb, DIMC, qb, DIMC,
        NROWS, DIMC, DIMC, nullptr, nullptr, 0);
    k_cross_q<<<NROWS, 256, 0, stream>>>(qb, qc);
    k_transp<<<dim3(DIMC/32, 2*DIMC/32), 256, 0, stream>>>(cwkv, DIMC, 2*DIMC, wtb);
    k_gemm<<<dim3(2*DIMC/64, (int)CTX/64), 128, 0, stream>>>(retbf, DIMC, wtb, DIMC,
        ckv, 2*DIMC, (int)CTX, 2*DIMC, DIMC, nullptr, nullptr, 0);
    k_cross_kv<<<NCHUNK*KVPAD, 256, 0, stream>>>(ckv, cnul, kck, kcv);
    k_flash<<<dim3(64/16, NHEAD, NCHUNK), 32, 0, stream>>>(qc, DIMC, kck, kcv, DIMC,
        abf, DIMC, 64, NR*NRC + 1, KVPAD, 0, cmask, NR*NRC);
    k_transp<<<dim3(DIMC/32, DIMC/32), 256, 0, stream>>>(cwo, DIMC, DIMC, wtb);
    k_gemm<<<dim3(DIMC/64, NROWS/64), 128, 0, stream>>>(abf, DIMC, wtb, DIMC, pj, DIMC,
        NROWS, DIMC, DIMC, cbo, nullptr, 1);
    k_scatter<<<NROWS, 256, 0, stream>>>(xb, pj);

    // ---- feed-forward ----
    k_rmsnorm<<<NROWS, 256, 0, stream>>>(xb, ff_g + (size_t)l * DIMC, nullptr, hbf);
    k_transp<<<dim3(DIMC/32, FFD/32), 256, 0, stream>>>(w1, DIMC, FFD, wtb);
    k_gemm<<<dim3(FFD/64, NROWS/64), 128, 0, stream>>>(hbf, DIMC, wtb, DIMC, ffbf, FFD,
        NROWS, FFD, DIMC, b1, nullptr, 1 | 2 | 8);
    k_transp<<<dim3(FFD/32, DIMC/32), 256, 0, stream>>>(w2, FFD, DIMC, wtb);
    k_gemm<<<dim3(DIMC/64, NROWS/64), 128, 0, stream>>>(ffbf, FFD, wtb, FFD, xb, DIMC,
        NROWS, DIMC, FFD, b2, xb, 1 | 4);
  }

  k_rmsnorm<<<NROWS, 256, 0, stream>>>(xb, out_g, (float*)d_out, nullptr);
}
